// SlidingWindowAttention_12592844112131
// MI455X (gfx1250) — matrix operand order repaired and checked by reading, not compiled, not run on hardware
//
#include <hip/hip_runtime.h>
#include <hip/hip_bf16.h>

typedef __bf16 bf16;
typedef __attribute__((ext_vector_type(16))) __bf16 v16bf;
typedef __attribute__((ext_vector_type(8)))  __bf16 v8bf;
typedef __attribute__((ext_vector_type(8)))  float   v8f;

#define NHEADS 16
#define SEQ    4096
#define CDIM   1024
#define HDIM   64
#define WINH   256
#define LDP    40   // padded LDS row stride in bf16 elems (32 data + 8 pad) -> conflict-free b128 reads

__device__ __forceinline__ v16bf cat8(v8bf lo, v8bf hi) {
  return __builtin_shufflevector(lo, hi, 0,1,2,3,4,5,6,7,8,9,10,11,12,13,14,15);
}
__device__ __forceinline__ v8f wmma_bf16(v16bf a, v16bf b, v8f c) {
  // D = A(16x32) * B(32x16) + C, f32 accumulate
  return __builtin_amdgcn_wmma_f32_16x16x32_bf16(false, a, false, b, (short)0, c, false, false);
}

// CDNA5 async copy: per-lane 16B global -> LDS, tracked by ASYNCcnt.
// lds_off = low 32 bits of flat LDS address (aperture rule: addr[31:0] = LDS offset).
__device__ __forceinline__ void async_cp16(unsigned lds_off, const bf16* g) {
  asm volatile("global_load_async_to_lds_b128 %0, %1, off"
               :: "v"(lds_off), "v"((unsigned long long)(uintptr_t)g) : "memory");
}
__device__ __forceinline__ void wait_async0() {
  asm volatile("s_wait_asynccnt 0x0" ::: "memory");
}
__device__ __forceinline__ unsigned lds_off(const bf16* p) {
  return (unsigned)(uintptr_t)p;
}

// ---------------------------------------------------------------------------
// f32 -> bf16 convert (one float4 -> 8B per thread). Grid sized exactly.
// ---------------------------------------------------------------------------
__global__ __launch_bounds__(256) void cvt_kernel(const float* __restrict__ s,
                                                  bf16* __restrict__ d) {
  size_t i = (size_t)blockIdx.x * 256 + threadIdx.x;
  float4 a = ((const float4*)s)[i];
  union { uint2 u; bf16 h[4]; } p;
  p.h[0] = (bf16)a.x; p.h[1] = (bf16)a.y; p.h[2] = (bf16)a.z; p.h[3] = (bf16)a.w;
  ((uint2*)d)[i] = p.u;
}

// ---------------------------------------------------------------------------
// GEMM1: qkv = x @ w_qkv^T (both pre-converted bf16, row-major [*,1024]).
// Async double-buffered LDS staging. Block tile 128x128, 8 waves (2m x 4n),
// each wave 64x32 = 4x2 accum frags, 8 WMMAs per 32-K step.
// Output: Q,K bf16 [H][N][D]; V transposed into VT [H][D][N].
// ---------------------------------------------------------------------------
__global__ __launch_bounds__(256) void qkv_gemm_kernel(
    const bf16* __restrict__ xb, const bf16* __restrict__ wb,
    bf16* __restrict__ Q, bf16* __restrict__ K, bf16* __restrict__ VT) {
  __shared__ __align__(16) bf16 sA[2][128 * LDP];
  __shared__ __align__(16) bf16 sB[2][128 * LDP];
  const int tid = threadIdx.x;
  const int mb = blockIdx.x;           // 0..31  (token tiles)
  const int nb = blockIdx.y;           // 0..23  (qkv-column tiles)
  const int wid = tid >> 5, lane = tid & 31;
  const int l15 = lane & 15, lh = lane >> 4;
  const int wm = wid >> 2, wn = wid & 3;

  // staging: 128x32 bf16 tile = 512 chunks of 8 bf16; thread covers chunk tid
  // and tid+256 -> rows r0 and r0+64, same column offset.
  const int r0 = tid >> 2, c0 = (tid & 3) * 8;
  const bf16* gA0 = xb + (size_t)(mb * 128 + r0) * CDIM + c0;
  const bf16* gA1 = gA0 + (size_t)64 * CDIM;
  const bf16* gB0 = wb + (size_t)(nb * 128 + r0) * CDIM + c0;
  const bf16* gB1 = gB0 + (size_t)64 * CDIM;
  unsigned la[2], lb[2];
  la[0] = lds_off(&sA[0][r0 * LDP + c0]);
  la[1] = lds_off(&sA[1][r0 * LDP + c0]);
  lb[0] = lds_off(&sB[0][r0 * LDP + c0]);
  lb[1] = lds_off(&sB[1][r0 * LDP + c0]);
  const unsigned dR = 64 * LDP * sizeof(bf16);   // LDS byte delta for row+64

  const v8f vz = {0.f,0.f,0.f,0.f,0.f,0.f,0.f,0.f};
  v8f acc[4][2];
#pragma unroll
  for (int i = 0; i < 4; ++i) { acc[i][0] = vz; acc[i][1] = vz; }

  // prologue: fill buffer 0
  async_cp16(la[0], gA0); async_cp16(la[0] + dR, gA1);
  async_cp16(lb[0], gB0); async_cp16(lb[0] + dR, gB1);
  gA0 += 32; gA1 += 32; gB0 += 32; gB1 += 32;

  for (int it = 0; it < 32; ++it) {
    const int b = it & 1;
    wait_async0();
    __syncthreads();
    if (it != 31) {                      // issue next tile while computing
      const int nbuf = b ^ 1;
      async_cp16(la[nbuf], gA0); async_cp16(la[nbuf] + dR, gA1);
      async_cp16(lb[nbuf], gB0); async_cp16(lb[nbuf] + dR, gB1);
      gA0 += 32; gA1 += 32; gB0 += 32; gB1 += 32;
    }
    const bf16* pA = sA[b];
    const bf16* pB = sB[b];
    v16bf af[4], bfr[2];
#pragma unroll
    for (int mf = 0; mf < 4; ++mf) {     // A frag: row m=l15, K = 16h..16h+15 contiguous
      const bf16* p = pA + (wm * 64 + mf * 16 + l15) * LDP;
      af[mf] = cat8(*(const v8bf*)(p + 16 * lh), *(const v8bf*)(p + 16 * lh + 8));
    }
#pragma unroll
    for (int nf = 0; nf < 2; ++nf) {     // B frag: col n=l15, K = 16h..16h+15 contiguous
      const bf16* p = pB + (wn * 32 + nf * 16 + l15) * LDP + 16 * lh;
      bfr[nf] = cat8(*(const v8bf*)p, *(const v8bf*)(p + 8));
    }
#pragma unroll
    for (int mf = 0; mf < 4; ++mf)
#pragma unroll
      for (int nf = 0; nf < 2; ++nf)
        acc[mf][nf] = wmma_bf16(af[mf], bfr[nf], acc[mf][nf]);
  }

  // Epilogue: C-frag lane holds (M = 8*lh + r, N = l15). Column c -> (s, head, d).
#pragma unroll
  for (int mf = 0; mf < 4; ++mf) {
    int token0 = mb * 128 + wm * 64 + mf * 16 + 8 * lh;   // 8 consecutive tokens per lane
#pragma unroll
    for (int nf = 0; nf < 2; ++nf) {
      int cbase = nb * 128 + wn * 32 + nf * 16;   // 16-aligned: s, head, dbase uniform in frag
      int s = cbase >> 10;
      int hh = (cbase >> 6) & 15;
      int dbase = cbase & 63;
      if (s == 2) {
        // V transposed: VT[(h*64+d)*SEQ + token]; 8 consecutive tokens -> one b128 store
        union { uint4 u; bf16 h[8]; } pk;
#pragma unroll
        for (int r = 0; r < 8; ++r) pk.h[r] = (bf16)acc[mf][nf][r];
        *(uint4*)(VT + (size_t)(hh * HDIM + dbase + l15) * SEQ + token0) = pk.u;
      } else {
        bf16* dst = (s == 0) ? Q : K;
#pragma unroll
        for (int r = 0; r < 8; ++r)
          dst[((size_t)hh * SEQ + token0 + r) * HDIM + dbase + l15] = (bf16)acc[mf][nf][r];
      }
    }
  }
}

// ---------------------------------------------------------------------------
// Banded flash attention: 1 wave = (head, 16-query tile), online softmax over
// 32-key steps inside the +/-256 window. 8 WMMAs per step (4 QK^T + 4 PV).
// ---------------------------------------------------------------------------
__global__ __launch_bounds__(256) void attn_kernel(
    const bf16* __restrict__ Q, const bf16* __restrict__ K,
    const bf16* __restrict__ VT, bf16* __restrict__ O) {
  __shared__ __align__(16) bf16 lP[8][16 * LDP];   // per-wave P bounce buffer
  const int tid = threadIdx.x;
  const int wid = tid >> 5, lane = tid & 31;
  const int l15 = lane & 15, lh = lane >> 4;
  const int wg = blockIdx.x * 8 + wid;
  const int head = wg >> 8;
  const int qbase = (wg & 255) * 16;
  const float scale = 0.125f;                      // 1/sqrt(64)
  const float NEGINF = -__builtin_inff();

  const bf16* qh = Q + (size_t)head * SEQ * HDIM;
  const bf16* kh = K + (size_t)head * SEQ * HDIM;
  const bf16* vh = VT + (size_t)head * HDIM * SEQ;

  v16bf qf[2];                                     // Q A-frags, d chunks 0..31 / 32..63
#pragma unroll
  for (int dc = 0; dc < 2; ++dc) {
    const bf16* p = qh + (size_t)(qbase + l15) * HDIM + dc * 32;
    qf[dc] = cat8(*(const v8bf*)(p + 16 * lh), *(const v8bf*)(p + 16 * lh + 8));
  }

  const v8f vz = {0.f,0.f,0.f,0.f,0.f,0.f,0.f,0.f};
  float mrow[8], lrow[8];
  v8f oacc[4];
#pragma unroll
  for (int r = 0; r < 8; ++r) { mrow[r] = NEGINF; lrow[r] = 0.f; }
#pragma unroll
  for (int f = 0; f < 4; ++f) oacc[f] = vz;

  int kb0 = qbase - WINH; if (kb0 < 0) kb0 = 0; kb0 &= ~31;
  int kbend = qbase + 16 + WINH; if (kbend > SEQ) kbend = SEQ;

  for (int kb = kb0; kb < kbend; kb += 32) {
    // prefetch next key block into WGP$ while this iteration computes
    if (kb + 32 < kbend) {
      __builtin_prefetch(kh + (size_t)(kb + 32 + lane) * HDIM, 0, 3);
      __builtin_prefetch(vh + (size_t)lane * SEQ + kb + 32, 0, 3);
      __builtin_prefetch(vh + (size_t)(32 + lane) * SEQ + kb + 32, 0, 3);
    }
    // S = Q K^T (contract d=64 in two 32-chunks), two 16-key sub-blocks
    v8f s0 = vz, s1 = vz;
#pragma unroll
    for (int dc = 0; dc < 2; ++dc) {
      const bf16* p0 = kh + (size_t)(kb + l15) * HDIM + dc * 32 + 16 * lh;
      const bf16* p1 = kh + (size_t)(kb + 16 + l15) * HDIM + dc * 32 + 16 * lh;
      v16bf kf0 = cat8(*(const v8bf*)p0, *(const v8bf*)(p0 + 8));
      v16bf kf1 = cat8(*(const v8bf*)p1, *(const v8bf*)(p1 + 8));
      s0 = wmma_bf16(qf[dc], kf0, s0);
      s1 = wmma_bf16(qf[dc], kf1, s1);
    }
    // mask + online softmax; row m = 8*lh + r lives on lanes [16*lh, 16*lh+15]
    float corr[8];
#pragma unroll
    for (int r = 0; r < 8; ++r) {
      int qi = qbase + 8 * lh + r;
      int d0 = qi - (kb + l15);
      int d1 = d0 - 16;
      float v0 = (d0 <= WINH && d0 >= -WINH) ? s0[r] * scale : NEGINF;
      float v1 = (d1 <= WINH && d1 >= -WINH) ? s1[r] * scale : NEGINF;
      float mx = fmaxf(v0, v1);
#pragma unroll
      for (int off = 1; off < 16; off <<= 1) mx = fmaxf(mx, __shfl_xor(mx, off, 32));
      float mnew = fmaxf(mrow[r], mx);               // always finite: every 32-key block hits window
      float p0 = __expf(v0 - mnew);
      float p1 = __expf(v1 - mnew);
      float sum = p0 + p1;
#pragma unroll
      for (int off = 1; off < 16; off <<= 1) sum += __shfl_xor(sum, off, 32);
      float c = __expf(mrow[r] - mnew);
      lrow[r] = lrow[r] * c + sum;
      mrow[r] = mnew;
      corr[r] = c;
      s0[r] = p0; s1[r] = p1;
    }
#pragma unroll
    for (int f = 0; f < 4; ++f)
#pragma unroll
      for (int r = 0; r < 8; ++r) oacc[f][r] *= corr[r];

    // P: C-layout -> A-layout via per-wave LDS (wave-synchronous, fence required)
    bf16* lp = lP[wid];
#pragma unroll
    for (int r = 0; r < 8; ++r) {
      int row = 8 * lh + r;
      lp[row * LDP + l15]      = (bf16)s0[r];
      lp[row * LDP + 16 + l15] = (bf16)s1[r];
    }
    asm volatile("s_wait_dscnt 0x0" ::: "memory");
    const bf16* pp = lp + l15 * LDP;
    v16bf pa = cat8(*(const v8bf*)(pp + 16 * lh), *(const v8bf*)(pp + 16 * lh + 8));

    // O += P * V  (V pre-transposed: B-frag = contiguous 32B per lane)
#pragma unroll
    for (int dc = 0; dc < 4; ++dc) {
      const bf16* p = vh + (size_t)(dc * 16 + l15) * SEQ + kb + 16 * lh;
      v16bf vf = cat8(*(const v8bf*)p, *(const v8bf*)(p + 8));
      oacc[dc] = wmma_bf16(pa, vf, oacc[dc]);
    }
  }

#pragma unroll
  for (int r = 0; r < 8; ++r) lrow[r] = 1.0f / lrow[r];
#pragma unroll
  for (int dc = 0; dc < 4; ++dc)
#pragma unroll
    for (int r = 0; r < 8; ++r) {
      int token = qbase + 8 * lh + r;
      O[(size_t)token * CDIM + head * HDIM + dc * 16 + l15] = (bf16)(oacc[dc][r] * lrow[r]);
    }
}

// ---------------------------------------------------------------------------
// GEMM2: out = attn_out(bf16) @ w_proj^T(bf16) + b_proj, f32 output.
// Same async double-buffered structure as GEMM1.
// ---------------------------------------------------------------------------
__global__ __launch_bounds__(256) void proj_gemm_kernel(
    const bf16* __restrict__ A, const bf16* __restrict__ W,
    const float* __restrict__ bias, float* __restrict__ out) {
  __shared__ __align__(16) bf16 sA[2][128 * LDP];
  __shared__ __align__(16) bf16 sB[2][128 * LDP];
  const int tid = threadIdx.x;
  const int mb = blockIdx.x;           // 0..31
  const int nb = blockIdx.y;           // 0..7
  const int wid = tid >> 5, lane = tid & 31;
  const int l15 = lane & 15, lh = lane >> 4;
  const int wm = wid >> 2, wn = wid & 3;

  const int r0 = tid >> 2, c0 = (tid & 3) * 8;
  const bf16* gA0 = A + (size_t)(mb * 128 + r0) * CDIM + c0;
  const bf16* gA1 = gA0 + (size_t)64 * CDIM;
  const bf16* gB0 = W + (size_t)(nb * 128 + r0) * CDIM + c0;
  const bf16* gB1 = gB0 + (size_t)64 * CDIM;
  unsigned la[2], lb[2];
  la[0] = lds_off(&sA[0][r0 * LDP + c0]);
  la[1] = lds_off(&sA[1][r0 * LDP + c0]);
  lb[0] = lds_off(&sB[0][r0 * LDP + c0]);
  lb[1] = lds_off(&sB[1][r0 * LDP + c0]);
  const unsigned dR = 64 * LDP * sizeof(bf16);

  const v8f vz = {0.f,0.f,0.f,0.f,0.f,0.f,0.f,0.f};
  v8f acc[4][2];
#pragma unroll
  for (int i = 0; i < 4; ++i) { acc[i][0] = vz; acc[i][1] = vz; }

  async_cp16(la[0], gA0); async_cp16(la[0] + dR, gA1);
  async_cp16(lb[0], gB0); async_cp16(lb[0] + dR, gB1);
  gA0 += 32; gA1 += 32; gB0 += 32; gB1 += 32;

  for (int it = 0; it < 32; ++it) {
    const int b = it & 1;
    wait_async0();
    __syncthreads();
    if (it != 31) {
      const int nbuf = b ^ 1;
      async_cp16(la[nbuf], gA0); async_cp16(la[nbuf] + dR, gA1);
      async_cp16(lb[nbuf], gB0); async_cp16(lb[nbuf] + dR, gB1);
      gA0 += 32; gA1 += 32; gB0 += 32; gB1 += 32;
    }
    const bf16* pA = sA[b];
    const bf16* pB = sB[b];
    v16bf af[4], bfr[2];
#pragma unroll
    for (int mf = 0; mf < 4; ++mf) {
      const bf16* p = pA + (wm * 64 + mf * 16 + l15) * LDP;
      af[mf] = cat8(*(const v8bf*)(p + 16 * lh), *(const v8bf*)(p + 16 * lh + 8));
    }
#pragma unroll
    for (int nf = 0; nf < 2; ++nf) {
      const bf16* p = pB + (wn * 32 + nf * 16 + l15) * LDP + 16 * lh;
      bfr[nf] = cat8(*(const v8bf*)p, *(const v8bf*)(p + 8));
    }
#pragma unroll
    for (int mf = 0; mf < 4; ++mf)
#pragma unroll
      for (int nf = 0; nf < 2; ++nf)
        acc[mf][nf] = wmma_bf16(af[mf], bfr[nf], acc[mf][nf]);
  }

#pragma unroll
  for (int nf = 0; nf < 2; ++nf) {
    int n = nb * 128 + wn * 32 + nf * 16 + l15;
    float bv = bias[n];
#pragma unroll
    for (int mf = 0; mf < 4; ++mf) {
      int token0 = mb * 128 + wm * 64 + mf * 16 + 8 * lh;
#pragma unroll
      for (int r = 0; r < 8; ++r)
        out[(size_t)(token0 + r) * CDIM + n] = acc[mf][nf][r] + bv;
    }
  }
}

extern "C" void kernel_launch(void* const* d_in, const int* in_sizes, int n_in,
                              void* d_out, int out_size, void* d_ws, size_t ws_size,
                              hipStream_t stream) {
  (void)in_sizes; (void)n_in; (void)out_size; (void)ws_size;
  const float* x      = (const float*)d_in[0];
  const float* w_qkv  = (const float*)d_in[1];
  const float* w_proj = (const float*)d_in[2];
  const float* b_proj = (const float*)d_in[3];

  char* ws = (char*)d_ws;
  const size_t MB = 1024 * 1024;
  bf16* Q   = (bf16*)(ws);            //  8 MB  [H][N][D]
  bf16* K   = (bf16*)(ws +  8 * MB);  //  8 MB  [H][N][D]
  bf16* VT  = (bf16*)(ws + 16 * MB);  //  8 MB  [H][D][N]
  bf16* AO  = (bf16*)(ws + 24 * MB);  //  8 MB  [N][C]
  bf16* XB  = (bf16*)(ws + 32 * MB);  //  8 MB  x in bf16
  bf16* WQB = (bf16*)(ws + 40 * MB);  //  6 MB  w_qkv in bf16
  bf16* WPB = (bf16*)(ws + 46 * MB);  //  2 MB  w_proj in bf16

  cvt_kernel<<<dim3(4096), 256, 0, stream>>>(x,      XB);
  cvt_kernel<<<dim3(3072), 256, 0, stream>>>(w_qkv,  WQB);
  cvt_kernel<<<dim3(1024), 256, 0, stream>>>(w_proj, WPB);
  qkv_gemm_kernel<<<dim3(32, 24), 256, 0, stream>>>(XB, WQB, Q, K, VT);
  attn_kernel<<<dim3(512), 256, 0, stream>>>(Q, K, VT, AO);
  proj_gemm_kernel<<<dim3(32, 8), 256, 0, stream>>>(AO, WPB, b_proj, (float*)d_out);
}
